// TransformerModel_75471165325810
// MI455X (gfx1250) — matrix operand order repaired and checked by reading, not compiled, not run on hardware
//
#include <hip/hip_runtime.h>

typedef __attribute__((ext_vector_type(16))) _Float16     v16h;
typedef __attribute__((ext_vector_type(8)))  _Float16     v8h;
typedef __attribute__((ext_vector_type(8)))  float        v8f;
typedef __attribute__((ext_vector_type(4)))  unsigned int u32x4;
typedef __attribute__((ext_vector_type(8)))  int          i32x8;
typedef __attribute__((ext_vector_type(4)))  int          i32x4;

// Model dims
static constexpr int Bb  = 16;
static constexpr int Ss  = 2048;
static constexpr int IND = 32;
static constexpr int Dd  = 512;
static constexpr int Ll  = 6;
static constexpr int FFd = 2048;
static constexpr int Mtok = Bb * Ss;                       // 32768 tokens
static constexpr int KC  = 256;                            // k-chunk (halves) staged per TDM op
static constexpr float SQRT_D     = 22.62741699796952f;    // sqrt(512)
static constexpr float INV_SQRT_D = 0.04419417382415922f;  // 1/sqrt(512)

__device__ __forceinline__ float wave_reduce_sum(float v) {
#pragma unroll
  for (int off = 16; off >= 1; off >>= 1) v += __shfl_xor(v, off, 32);
  return v;
}

// ---------------- fp32 -> fp16 weight conversion ----------------
__global__ __launch_bounds__(256) void cvt_f16_kernel(const float* __restrict__ in,
                                                      _Float16* __restrict__ out, int n) {
  int i = blockIdx.x * 256 + threadIdx.x;
  if (i < n) out[i] = (_Float16)in[i];
}

// ---------------- encoder: x = (src @ enc_w.T + b)*sqrt(D) + pe[batch] ----------------
__global__ __launch_bounds__(256) void encoder_kernel(const float* __restrict__ src,
                                                      const float* __restrict__ pe,
                                                      const float* __restrict__ ew,
                                                      const float* __restrict__ eb,
                                                      float* __restrict__ x) {
  const int t = blockIdx.x;          // token id 0..32767
  const int b = t >> 11;             // pe indexed by BATCH (matches reference's pe[:B])
  const float* srow = src + (size_t)t * IND;
  float sv[IND];
#pragma unroll
  for (int k = 0; k < IND; ++k) sv[k] = srow[k];
  for (int d = threadIdx.x; d < Dd; d += 256) {
    const float* wrow = ew + (size_t)d * IND;
    float acc = eb[d];
#pragma unroll
    for (int k = 0; k < IND; ++k) acc += sv[k] * wrow[k];
    x[(size_t)t * Dd + d] = acc * SQRT_D + pe[(size_t)b * Dd + d];
  }
}

// ---------------- windowed attention (W=3) + residual + LayerNorm1 ----------------
__global__ __launch_bounds__(256) void attn_ln_kernel(const float* __restrict__ x,
                                                      const float* __restrict__ gam,
                                                      const float* __restrict__ bet,
                                                      float* __restrict__ y,
                                                      _Float16* __restrict__ yh) {
  const int wave = threadIdx.x >> 5;
  const int lane = threadIdx.x & 31;
  const int t = blockIdx.x * 8 + wave;       // token id
  const int b = t >> 11;
  const int s = t & (Ss - 1);
  const int d0 = lane * 16;
  const float* base = x + (size_t)b * Ss * Dd;
  const float* qp = base + (size_t)s * Dd + d0;

  float q[16];
#pragma unroll
  for (int i = 0; i < 16; ++i) q[i] = qp[i];

  float sc[7];
#pragma unroll
  for (int j = 0; j < 7; ++j) {
    const int idx = s + j - 3;
    const int ki = idx < 0 ? 0 : (idx > Ss - 1 ? Ss - 1 : idx);
    const float* kp = base + (size_t)ki * Dd + d0;
    float p = 0.f;
#pragma unroll
    for (int i = 0; i < 16; ++i) p += q[i] * kp[i];
    p = wave_reduce_sum(p);
    sc[j] = (idx >= 0 && idx < Ss) ? p * INV_SQRT_D : -1.0e9f;
  }
  float mx = sc[0];
#pragma unroll
  for (int j = 1; j < 7; ++j) mx = fmaxf(mx, sc[j]);
  float den = 0.f;
#pragma unroll
  for (int j = 0; j < 7; ++j) { sc[j] = __expf(sc[j] - mx); den += sc[j]; }
  const float inv = 1.f / den;

  float o[16];
#pragma unroll
  for (int i = 0; i < 16; ++i) o[i] = 0.f;
#pragma unroll
  for (int j = 0; j < 7; ++j) {
    const int idx = s + j - 3;
    const int ki = idx < 0 ? 0 : (idx > Ss - 1 ? Ss - 1 : idx);
    const float* vp = base + (size_t)ki * Dd + d0;
    const float w = sc[j] * inv;
#pragma unroll
    for (int i = 0; i < 16; ++i) o[i] += w * vp[i];
  }

  float r[16], ssum = 0.f, ssq = 0.f;
#pragma unroll
  for (int i = 0; i < 16; ++i) { r[i] = q[i] + o[i]; ssum += r[i]; ssq += r[i] * r[i]; }
  ssum = wave_reduce_sum(ssum);
  ssq  = wave_reduce_sum(ssq);
  const float mu = ssum * (1.f / Dd);
  const float var = ssq * (1.f / Dd) - mu * mu;
  const float rstd = rsqrtf(var + 1e-5f);

  float* yp = y + (size_t)t * Dd + d0;
  _Float16* yhp = yh + (size_t)t * Dd + d0;
#pragma unroll
  for (int i = 0; i < 16; ++i) {
    const float v = (r[i] - mu) * rstd * gam[d0 + i] + bet[d0 + i];
    yp[i] = v;
    yhp[i] = (_Float16)v;
  }
}

// ---------------- WMMA GEMM with TDM-staged B tile ----------------
// C[M,N] = act(A[M,K] @ W[N,K]^T + bias)
// grid = (N/64, M/128), block = 256 (8 waves). All 8 waves share one 64-row B strip:
// wave 0 DMA-stages it into LDS in KC-chunks via tensor_load_to_lds (double buffered,
// synchronized with s_wait_tensorcnt + barriers); waves read B frags with ds_load.
template <int K, int N, bool RELU, typename OUTT>
__global__ __launch_bounds__(256) void gemm_bias_kernel(const _Float16* __restrict__ A,
                                                        const _Float16* __restrict__ W,
                                                        const float* __restrict__ bias,
                                                        OUTT* __restrict__ C) {
  static_assert(K % KC == 0, "K must be a multiple of KC");
  constexpr int NCH = K / KC;

  __shared__ __attribute__((aligned(32))) _Float16 bsm[2][64 * KC];

  const int wave = threadIdx.x >> 5;
  const int lane = threadIdx.x & 31;
  const int half = lane >> 4;     // 0 or 1
  const int l16  = lane & 15;
  const int mbase = (blockIdx.y * 8 + wave) * 16;
  const int nbase = blockIdx.x * 64;

  v8f acc[4];
#pragma unroll
  for (int nt = 0; nt < 4; ++nt)
#pragma unroll
    for (int i = 0; i < 8; ++i) acc[nt][i] = 0.f;

  const _Float16* arow = A + (size_t)(mbase + l16) * K;

#if __has_builtin(__builtin_amdgcn_tensor_load_to_lds)
  // Issue one TDM 2-D tile load: 64 rows x KC halves of W starting at (nbase, c*KC)
  auto tdm_issue = [&](int c, int buf) {
    const unsigned long long ga =
        (unsigned long long)(uintptr_t)(W + (size_t)nbase * K + (size_t)c * KC);
    const unsigned int lds = (unsigned int)(uintptr_t)(&bsm[buf][0]);
    u32x4 g0;
    g0[0] = 1u;                                   // count=1 (valid), user mode
    g0[1] = lds;                                  // lds_addr [63:32]
    g0[2] = (unsigned int)ga;                     // global_addr [95:64]
    g0[3] = (unsigned int)((ga >> 32) & 0x1FFFFFFu) | (2u << 30);  // addr hi + type=2
    i32x8 g1;
    g1[0] = (1 << 16);        // data_size = 2 bytes; no mask/iterate/pad
    g1[1] = (KC << 16);       // tensor_dim0 = KC  [79:48] (low 16)
    g1[2] = (64 << 16);       // tensor_dim0 hi=0, tensor_dim1 = 64 [111:80] (low 16)
    g1[3] = (KC << 16);       // tensor_dim1 hi=0, tile_dim0 = KC [127:112]
    g1[4] = 64;               // tile_dim1 = 64, tile_dim2 = 0
    g1[5] = K;                // tensor_dim0_stride = K [207:160] (low 32)
    g1[6] = 0;                // stride hi=0, tensor_dim1_stride lo=0
    g1[7] = 0;
    i32x4 g2 = {0, 0, 0, 0};
    i32x4 g3 = {0, 0, 0, 0};
#if __clang_major__ >= 23
    i32x8 g4 = {0, 0, 0, 0, 0, 0, 0, 0};
    __builtin_amdgcn_tensor_load_to_lds(g0, g1, g2, g3, g4, 0);
#else
    __builtin_amdgcn_tensor_load_to_lds(g0, g1, g2, g3, 0);
#endif
  };

  if (threadIdx.x < 32) tdm_issue(0, 0);

  for (int c = 0; c < NCH; ++c) {
    if (threadIdx.x < 32) {
      if (c + 1 < NCH) {
        tdm_issue(c + 1, (c + 1) & 1);
        __builtin_amdgcn_s_wait_tensorcnt((short)1);   // chunk c landed, c+1 in flight
      } else {
        __builtin_amdgcn_s_wait_tensorcnt((short)0);
      }
    }
    __syncthreads();   // B chunk c visible to all waves

    if (c + 1 < NCH) __builtin_prefetch(arow + (size_t)(c + 1) * KC, 0, 0);

    const _Float16* bbuf = &bsm[c & 1][0];
#pragma unroll
    for (int kk = 0; kk < KC; kk += 32) {
      const int kg = c * KC + kk;
      const v8h alo = *(const v8h*)(arow + kg + 16 * half);
      const v8h ahi = *(const v8h*)(arow + kg + 16 * half + 8);
      v16h a;
#pragma unroll
      for (int i = 0; i < 8; ++i) { a[i] = alo[i]; a[8 + i] = ahi[i]; }
#pragma unroll
      for (int nt = 0; nt < 4; ++nt) {
        const v16h bfrag = *(const v16h*)(bbuf + (nt * 16 + l16) * KC + kk + 16 * half);
        acc[nt] = __builtin_amdgcn_wmma_f32_16x16x32_f16(
            false, a, false, bfrag, (short)0, acc[nt], false, false);
      }
    }
    __syncthreads();   // all waves done with buf[c&1] before it is re-filled
  }
#else
  // Fallback: cooperative global->LDS staging with plain loads + ds stores
  for (int c = 0; c < NCH; ++c) {
    __syncthreads();
    for (int e = threadIdx.x; e < 64 * (KC / 8); e += 256) {
      const int row = e / (KC / 8);
      const int kv  = (e % (KC / 8)) * 8;
      *(v8h*)(&bsm[c & 1][row * KC + kv]) =
          *(const v8h*)(W + (size_t)(nbase + row) * K + c * KC + kv);
    }
    __syncthreads();
    const _Float16* bbuf = &bsm[c & 1][0];
#pragma unroll
    for (int kk = 0; kk < KC; kk += 32) {
      const int kg = c * KC + kk;
      const v8h alo = *(const v8h*)(arow + kg + 16 * half);
      const v8h ahi = *(const v8h*)(arow + kg + 16 * half + 8);
      v16h a;
#pragma unroll
      for (int i = 0; i < 8; ++i) { a[i] = alo[i]; a[8 + i] = ahi[i]; }
#pragma unroll
      for (int nt = 0; nt < 4; ++nt) {
        const v16h bfrag = *(const v16h*)(bbuf + (nt * 16 + l16) * KC + kk + 16 * half);
        acc[nt] = __builtin_amdgcn_wmma_f32_16x16x32_f16(
            false, a, false, bfrag, (short)0, acc[nt], false, false);
      }
    }
  }
#endif

#pragma unroll
  for (int nt = 0; nt < 4; ++nt) {
    const int col = nbase + nt * 16 + l16;
    const float bv = bias[col];
#pragma unroll
    for (int r = 0; r < 8; ++r) {
      const int row = mbase + r + 8 * half;
      float v = acc[nt][r] + bv;
      if (RELU) v = fmaxf(v, 0.f);
      C[(size_t)row * N + col] = (OUTT)v;
    }
  }
}

// ---------------- residual + LayerNorm2: x = LN(y + f) ----------------
__global__ __launch_bounds__(256) void add_ln_kernel(const float* __restrict__ y,
                                                     const float* __restrict__ f,
                                                     const float* __restrict__ gam,
                                                     const float* __restrict__ bet,
                                                     float* __restrict__ xo) {
  const int wave = threadIdx.x >> 5;
  const int lane = threadIdx.x & 31;
  const int t = blockIdx.x * 8 + wave;
  const int d0 = lane * 16;
  const float* yp = y + (size_t)t * Dd + d0;
  const float* fp = f + (size_t)t * Dd + d0;
  float r[16], ssum = 0.f, ssq = 0.f;
#pragma unroll
  for (int i = 0; i < 16; ++i) { r[i] = yp[i] + fp[i]; ssum += r[i]; ssq += r[i] * r[i]; }
  ssum = wave_reduce_sum(ssum);
  ssq  = wave_reduce_sum(ssq);
  const float mu = ssum * (1.f / Dd);
  const float var = ssq * (1.f / Dd) - mu * mu;
  const float rstd = rsqrtf(var + 1e-5f);
  float* xp = xo + (size_t)t * Dd + d0;
#pragma unroll
  for (int i = 0; i < 16; ++i)
    xp[i] = (r[i] - mu) * rstd * gam[d0 + i] + bet[d0 + i];
}

// ---------------- decode: out[b] = x[b, S-1, :] . dec_w + dec_b ----------------
__global__ __launch_bounds__(512) void decode_kernel(const float* __restrict__ x,
                                                     const float* __restrict__ dw,
                                                     const float* __restrict__ db,
                                                     float* __restrict__ out) {
  const int wave = threadIdx.x >> 5;   // batch index (16 waves)
  const int lane = threadIdx.x & 31;
  const float* xp = x + ((size_t)wave * Ss + (Ss - 1)) * Dd + lane * 16;
  const float* wp = dw + lane * 16;
  float p = 0.f;
#pragma unroll
  for (int i = 0; i < 16; ++i) p += xp[i] * wp[i];
  p = wave_reduce_sum(p);
  if (lane == 0) out[wave] = p + db[0];
}

extern "C" void kernel_launch(void* const* d_in, const int* in_sizes, int n_in,
                              void* d_out, int out_size, void* d_ws, size_t ws_size,
                              hipStream_t stream) {
  (void)in_sizes; (void)n_in; (void)out_size; (void)ws_size;
  const float* src   = (const float*)d_in[0];
  const float* pe    = (const float*)d_in[1];
  const float* enc_w = (const float*)d_in[2];
  const float* enc_b = (const float*)d_in[3];
  const float* l1_w  = (const float*)d_in[4];
  const float* l1_b  = (const float*)d_in[5];
  const float* l2_w  = (const float*)d_in[6];
  const float* l2_b  = (const float*)d_in[7];
  const float* n1_g  = (const float*)d_in[8];
  const float* n1_b  = (const float*)d_in[9];
  const float* n2_g  = (const float*)d_in[10];
  const float* n2_b  = (const float*)d_in[11];
  const float* dec_w = (const float*)d_in[12];
  const float* dec_b = (const float*)d_in[13];
  float* out = (float*)d_out;

  char* ws = (char*)d_ws;
  size_t off = 0;
  auto alloc = [&](size_t bytes) -> char* {
    char* p = ws + off;
    off += (bytes + 255) & ~(size_t)255;
    return p;
  };
  float*    xf  = (float*)   alloc((size_t)Mtok * Dd  * sizeof(float));
  float*    yf  = (float*)   alloc((size_t)Mtok * Dd  * sizeof(float));
  _Float16* yh  = (_Float16*)alloc((size_t)Mtok * Dd  * sizeof(_Float16));
  _Float16* Hh  = (_Float16*)alloc((size_t)Mtok * FFd * sizeof(_Float16));
  float*    Ff  = (float*)   alloc((size_t)Mtok * Dd  * sizeof(float));
  _Float16* w1h = (_Float16*)alloc((size_t)Ll * FFd * Dd * sizeof(_Float16));
  _Float16* w2h = (_Float16*)alloc((size_t)Ll * Dd * FFd * sizeof(_Float16));

  const int nW = Ll * FFd * Dd;  // 6,291,456 each
  cvt_f16_kernel<<<(nW + 255) / 256, 256, 0, stream>>>(l1_w, w1h, nW);
  cvt_f16_kernel<<<(nW + 255) / 256, 256, 0, stream>>>(l2_w, w2h, nW);

  encoder_kernel<<<Mtok, 256, 0, stream>>>(src, pe, enc_w, enc_b, xf);

  for (int l = 0; l < Ll; ++l) {
    attn_ln_kernel<<<Mtok / 8, 256, 0, stream>>>(
        xf, n1_g + (size_t)l * Dd, n1_b + (size_t)l * Dd, yf, yh);

    gemm_bias_kernel<Dd, FFd, true, _Float16>
        <<<dim3(FFd / 64, Mtok / 128), 256, 0, stream>>>(
            yh, w1h + (size_t)l * FFd * Dd, l1_b + (size_t)l * FFd, Hh);

    gemm_bias_kernel<FFd, Dd, false, float>
        <<<dim3(Dd / 64, Mtok / 128), 256, 0, stream>>>(
            Hh, w2h + (size_t)l * Dd * FFd, l2_b + (size_t)l * Dd, Ff);

    add_ln_kernel<<<Mtok / 8, 256, 0, stream>>>(
        yf, Ff, n2_g + (size_t)l * Dd, n2_b + (size_t)l * Dd, xf);
  }

  decode_kernel<<<1, 512, 0, stream>>>(xf, dec_w, dec_b, out);
}
